// FCOSPostprocessor_70952859729950
// MI455X (gfx1250) — matrix operand order repaired and checked by reading, not compiled, not run on hardware
//
#include <hip/hip_runtime.h>
#include <math.h>

// ---------------- constants ----------------
#define NEGF   (-1e9f)
#define NCLS   81
#define TOPN   1000
#define M3     3000           // 3 levels * 1000
#define KPAD   112            // 100 padded to 7 * 16
#define HWSEG  65536          // 256*256
#define EMB    32

typedef _Float16 v4h  __attribute__((ext_vector_type(4)));
typedef _Float16 v8h  __attribute__((ext_vector_type(8)));
typedef _Float16 v16h __attribute__((ext_vector_type(16)));
typedef float    v4f  __attribute__((ext_vector_type(4)));
typedef float    v8f  __attribute__((ext_vector_type(8)));

__device__ __forceinline__ float sigm(float x) { return 1.0f / (1.0f + __expf(-x)); }
__device__ __forceinline__ float clampf(float v, float lo, float hi) {
    return fminf(fmaxf(v, lo), hi);
}
// monotonic float -> uint key (larger float => larger key)
__device__ __forceinline__ unsigned keyOf(float x) {
    unsigned u = __float_as_uint(x);
    return (u & 0x80000000u) ? ~u : (u | 0x80000000u);
}

// ---------------- 0: init scratch ----------------
__global__ void init_kernel(unsigned* hist, unsigned* counters,
                            float* mScores, float* mBoxes, int* mLabels, int* mLocs,
                            int* keep) {
    int i = blockIdx.x * 256 + threadIdx.x;
    if (i < 3 * 2048) hist[i] = 0u;
    if (i < 3)        counters[i] = 0u;
    if (i < M3) { mScores[i] = NEGF; keep[i] = 0; mLabels[i] = 0; mLocs[i] = 0; }
    if (i < M3 * 4)   mBoxes[i] = 0.0f;
}

// ---------------- 1: sigmoid/mask + 2048-bin histogram (per level) ----------------
__global__ void score_hist_kernel(const float* __restrict__ cls,
                                  const float* __restrict__ ctr,
                                  float* __restrict__ masked,
                                  unsigned* __restrict__ hist, int N) {
    __shared__ unsigned lh[2048];
    for (int i = threadIdx.x; i < 2048; i += 256) lh[i] = 0u;
    __syncthreads();
    int total = N * NCLS;
    for (int i = blockIdx.x * 256 + threadIdx.x; i < total; i += gridDim.x * 256) {
        int n = i / NCLS, c = i - n * NCLS;
        float cs = sigm(cls[c * N + n]);
        float s = (cs > 0.05f) ? cs * sigm(ctr[n]) : NEGF;
        masked[i] = s;
        atomicAdd(&lh[keyOf(s) >> 21], 1u);
    }
    __syncthreads();
    for (int i = threadIdx.x; i < 2048; i += 256)
        if (lh[i]) atomicAdd(&hist[i], lh[i]);
}

// ---------------- 2: find per-level cutoff key (top-1000 bin) ----------------
__global__ void cutoff_kernel(const unsigned* __restrict__ hist, unsigned* cutoffKey) {
    int l = threadIdx.x;
    if (l < 3) {
        const unsigned* h = hist + l * 2048;
        unsigned cum = 0; int b = 0;
        for (int i = 2047; i >= 0; --i) { cum += h[i]; if (cum >= (unsigned)TOPN) { b = i; break; } }
        cutoffKey[l] = ((unsigned)b) << 21;
    }
}

// ---------------- 3: compact top candidates + decode boxes (per level) ----------------
__global__ void compact_kernel(const float* __restrict__ masked,
                               const float* __restrict__ box,
                               const float* __restrict__ loc,
                               const unsigned* __restrict__ cutoffKey,
                               unsigned* counters, int level,
                               float* mScores, float* mBoxes, int* mLabels, int* mLocs,
                               int N, int accOff,
                               const int* __restrict__ img_h_p,
                               const int* __restrict__ img_w_p) {
    int total = N * NCLS;
    int i = blockIdx.x * 256 + threadIdx.x;
    if (i >= total) return;
    float v = masked[i];
    if (keyOf(v) < cutoffKey[level]) return;
    unsigned slot = atomicAdd(&counters[level], 1u);
    if (slot >= (unsigned)TOPN) return;
    int n = i / NCLS, c = i - n * NCLS;
    float W = (float)img_w_p[0] - 1.0f;
    float H = (float)img_h_p[0] - 1.0f;
    float lx = loc[n * 2], ly = loc[n * 2 + 1];
    float b0 = box[0 * N + n], b1 = box[1 * N + n], b2 = box[2 * N + n], b3 = box[3 * N + n];
    float x1 = clampf(lx - b0, 0.0f, W);
    float y1 = clampf(ly - b1, 0.0f, H);
    float x2 = clampf(lx + b2, 0.0f, W);
    float y2 = clampf(ly + b3, 0.0f, H);
    bool valid = (v > NEGF * 0.5f) && (x2 - x1 >= 0.0f) && (y2 - y1 >= 0.0f);
    int o = level * TOPN + (int)slot;
    mScores[o] = valid ? sqrtf(fmaxf(v, 1e-12f)) : NEGF;
    mBoxes[o * 4 + 0] = x1; mBoxes[o * 4 + 1] = y1;
    mBoxes[o * 4 + 2] = x2; mBoxes[o * 4 + 3] = y2;
    mLabels[o] = c;
    mLocs[o] = n + accOff;
}

// ---------------- 4: per-class greedy NMS (block = one class) ----------------
__global__ void nms_kernel(const float* __restrict__ scores,
                           const float* __restrict__ boxes,
                           const int* __restrict__ labels,
                           int* keep) {
    int j = blockIdx.x + 1;   // classes 1..80
    __shared__ float live[M3];
    __shared__ float rv[256];
    __shared__ int   ri[256];
    __shared__ float bb[4];
    __shared__ float bval;
    int tid = threadIdx.x;
    for (int m = tid; m < M3; m += 256) live[m] = (labels[m] == j) ? scores[m] : NEGF;
    __syncthreads();
    for (int it = 0; it < 100; ++it) {
        float bv = -3e38f; int bi = 0;
        for (int m = tid; m < M3; m += 256) {
            float v = live[m];
            if (v > bv) { bv = v; bi = m; }
        }
        rv[tid] = bv; ri[tid] = bi;
        __syncthreads();
        for (int s = 128; s > 0; s >>= 1) {
            if (tid < s) {
                if (rv[tid + s] > rv[tid] || (rv[tid + s] == rv[tid] && ri[tid + s] < ri[tid])) {
                    rv[tid] = rv[tid + s]; ri[tid] = ri[tid + s];
                }
            }
            __syncthreads();
        }
        if (tid == 0) {
            int bidx = ri[0]; bval = rv[0];
            if (bval > NEGF * 0.5f) keep[bidx] = 1;
            bb[0] = boxes[bidx * 4 + 0]; bb[1] = boxes[bidx * 4 + 1];
            bb[2] = boxes[bidx * 4 + 2]; bb[3] = boxes[bidx * 4 + 3];
            live[bidx] = NEGF;
        }
        __syncthreads();
        if (bval > NEGF * 0.5f) {
            float a1 = fmaxf(bb[2] - bb[0], 0.0f) * fmaxf(bb[3] - bb[1], 0.0f);
            for (int m = tid; m < M3; m += 256) {
                float x1 = fmaxf(bb[0], boxes[m * 4 + 0]);
                float y1 = fmaxf(bb[1], boxes[m * 4 + 1]);
                float x2 = fminf(bb[2], boxes[m * 4 + 2]);
                float y2 = fminf(bb[3], boxes[m * 4 + 3]);
                float inter = fmaxf(x2 - x1, 0.0f) * fmaxf(y2 - y1, 0.0f);
                float a2 = fmaxf(boxes[m * 4 + 2] - boxes[m * 4 + 0], 0.0f) *
                           fmaxf(boxes[m * 4 + 3] - boxes[m * 4 + 1], 0.0f);
                float iou = inter / (a1 + a2 - inter + 1e-9f);
                if (iou > 0.6f) live[m] = NEGF;
            }
        }
        __syncthreads();
    }
}

// ---------------- 5: final top-100, write scalar outputs ----------------
__global__ void final_kernel(const float* __restrict__ scores,
                             const float* __restrict__ boxes,
                             const int* __restrict__ labels,
                             const int* __restrict__ locs,
                             const int* __restrict__ keep,
                             float* out, int* locp, int* okf) {
    __shared__ float live[M3];
    __shared__ float rv[256];
    __shared__ int   ri[256];
    int tid = threadIdx.x;
    for (int m = tid; m < M3; m += 256) live[m] = keep[m] ? scores[m] : NEGF;
    __syncthreads();
    for (int r = 0; r < 100; ++r) {
        float bv = -3e38f; int bi = 0;
        for (int m = tid; m < M3; m += 256) {
            float v = live[m];
            if (v > bv) { bv = v; bi = m; }
        }
        rv[tid] = bv; ri[tid] = bi;
        __syncthreads();
        for (int s = 128; s > 0; s >>= 1) {
            if (tid < s) {
                if (rv[tid + s] > rv[tid] || (rv[tid + s] == rv[tid] && ri[tid + s] < ri[tid])) {
                    rv[tid] = rv[tid + s]; ri[tid] = ri[tid + s];
                }
            }
            __syncthreads();
        }
        if (tid == 0) {
            int idx = ri[0]; float v = rv[0];
            bool ok = v > NEGF * 0.5f;
            out[r]        = ok ? v : -1.0f;
            out[100 + r]  = ok ? (float)labels[idx] : -1.0f;
            for (int c = 0; c < 4; ++c)
                out[200 + r * 4 + c] = ok ? (float)((int)boxes[idx * 4 + c]) : 0.0f;
            locp[r] = ok ? locs[idx] : 0;
            okf[r]  = ok ? 1 : 0;
            live[idx] = NEGF;
        }
        __syncthreads();
    }
}

// ---------------- 6: can_pred [E,HW] f32 -> canT [HW,E] f16, LDS-tiled transpose ----------------
// block handles a tile of 64 positions x 32 embeddings
__global__ void convert_can_kernel(const float* __restrict__ can, _Float16* __restrict__ canT) {
    __shared__ float tile[64][33];   // +1 pad: dodge LDS bank conflicts
    int pbase = blockIdx.x * 64;
    int t = threadIdx.x;
    int e0 = t >> 6;                 // 0..3
    int pl = t & 63;                 // 0..63
    // coalesced loads: 64 consecutive p per embedding row
    for (int ee = e0; ee < EMB; ee += 4)
        tile[pl][ee] = can[(size_t)ee * HWSEG + pbase + pl];
    __syncthreads();
    // coalesced stores: consecutive threads write consecutive 8-byte chunks
    for (int rr = 0; rr < 64; rr += 32) {
        int row = (t >> 3) + rr;     // 0..63
        int c4  = (t & 7) * 4;       // 0..28
        v4h h = { (_Float16)tile[row][c4 + 0], (_Float16)tile[row][c4 + 1],
                  (_Float16)tile[row][c4 + 2], (_Float16)tile[row][c4 + 3] };
        *(v4h*)(canT + (size_t)(pbase + row) * EMB + c4) = h;
    }
}

// ---------------- 7: gather selected ins vectors -> ins16 [KPAD,32] f16 ----------------
__global__ void gather_ins_kernel(const float* __restrict__ ins,
                                  const int* __restrict__ locp,
                                  const int* __restrict__ okf,
                                  _Float16* __restrict__ ins16) {
    int t = blockIdx.x * 256 + threadIdx.x;
    if (t >= KPAD * EMB) return;
    int k = t / EMB, e = t - k * EMB;
    float v = 0.0f;
    if (k < 100 && okf[k]) v = ins[(size_t)locp[k] * EMB + e];
    ins16[t] = (_Float16)v;
}

// ---------------- 8: WMMA GEMM seg[KPAD,HW] = ins16[KPAD,32] * canT^T ----------------
// one wave32 per 16x16 M-tile x 4 consecutive N-tiles; A fragment loaded once,
// 4 independent v_wmma_f32_16x16x32_f16 (no D->A/B hazards), B prefetched ahead.
#define NTPW 4
__global__ void wmma_gemm_kernel(const _Float16* __restrict__ ins16,
                                 const _Float16* __restrict__ canT,
                                 float* __restrict__ seg) {
    const int MT = KPAD / 16;         // 7
    const int NT = HWSEG / 16;        // 4096
    int wave = (blockIdx.x * 256 + threadIdx.x) >> 5;
    int lane = threadIdx.x & 31;
    int nWaves = MT * (NT / NTPW);
    if (wave >= nWaves) return;
    int mt  = wave % MT;
    int ntg = wave / MT;
    int hf = lane >> 4;               // half-wave
    int mr = lane & 15;

    // A fragment: row M = mt*16+mr; K runs {hf*8..hf*8+7} and {16+hf*8..16+hf*8+7}
    const _Float16* ap = ins16 + (size_t)(mt * 16 + mr) * EMB;
    union { v16h v; v8h h[2]; } au;
    au.h[0] = *(const v8h*)(ap + hf * 8);
    au.h[1] = *(const v8h*)(ap + 16 + hf * 8);
    int rowBase = mt * 16 + hf * 8;

    #pragma unroll
    for (int q = 0; q < NTPW; ++q) {
        int nt  = ntg * NTPW + q;
        int col = nt * 16 + mr;
        // B fragment: col N; K runs {hf*8..hf*8+7} and {16+hf*8..16+hf*8+7} (canT[p][e])
        const _Float16* bp = canT + (size_t)col * EMB + hf * 8;
        union { v16h v; v8h h[2]; } bu;
        bu.h[0] = *(const v8h*)(bp);
        bu.h[1] = *(const v8h*)(bp + 16);
        if (q + 1 < NTPW) __builtin_prefetch(bp + 16 * EMB, 0, 0);  // next tile's B
        v8f c = {};
        c = __builtin_amdgcn_wmma_f32_16x16x32_f16(false, au.v, false, bu.v,
                                                   (short)0, c, false, false);
        // D layout: VGPR r holds M = r + hf*8, N = lane%16
        #pragma unroll
        for (int r = 0; r < 8; ++r)
            seg[(size_t)(rowBase + r) * HWSEG + col] = c[r];
    }
}

// ---------------- 9: fused 4x bilinear upsample + sigmoid>0.5 threshold ----------------
// 419 MB write-once output (> 192 MB L2): non-temporal stores keep L2 for seg/canT.
__global__ void resize_kernel(const float* __restrict__ seg,
                              const int* __restrict__ okf,
                              float* __restrict__ out) {
    long long t = (long long)blockIdx.x * 256 + threadIdx.x;
    const long long TOT = 100LL * 1024 * 256;   // float4 granules
    if (t >= TOT) return;
    int xq = (int)(t & 255);
    int y  = (int)((t >> 8) & 1023);
    int k  = (int)(t >> 18);
    v4f r;
    if (!okf[k]) {
        r = (v4f){0.0f, 0.0f, 0.0f, 0.0f};
    } else {
        float sy = (y + 0.5f) * 0.25f - 0.5f;
        int y0; float wy;
        if (sy <= 0.0f)        { y0 = 0;   wy = 0.0f; }
        else if (sy >= 255.0f) { y0 = 254; wy = 1.0f; }
        else { y0 = (int)sy; if (y0 > 254) y0 = 254; wy = sy - (float)y0; }
        const float* s0 = seg + (size_t)k * HWSEG + (size_t)y0 * 256;
        const float* s1 = s0 + 256;
        #pragma unroll
        for (int j = 0; j < 4; ++j) {
            int x = xq * 4 + j;
            float sx = (x + 0.5f) * 0.25f - 0.5f;
            int x0; float wx;
            if (sx <= 0.0f)        { x0 = 0;   wx = 0.0f; }
            else if (sx >= 255.0f) { x0 = 254; wx = 1.0f; }
            else { x0 = (int)sx; if (x0 > 254) x0 = 254; wx = sx - (float)x0; }
            float top = s0[x0] * (1.0f - wx) + s0[x0 + 1] * wx;
            float bot = s1[x0] * (1.0f - wx) + s1[x0 + 1] * wx;
            float v = top * (1.0f - wy) + bot * wy;
            r[j] = (v > 0.0f) ? 1.0f : 0.0f;   // sigmoid(v) > 0.5
        }
    }
    float* dst = out + 600 + (size_t)k * (1024 * 1024) + (size_t)y * 1024 + xq * 4;
    __builtin_nontemporal_store(r, (v4f*)dst);
}

// ---------------- host launch ----------------
static inline size_t alignup(size_t x, size_t a) { return (x + a - 1) & ~(a - 1); }

extern "C" void kernel_launch(void* const* d_in, const int* in_sizes, int n_in,
                              void* d_out, int out_size, void* d_ws, size_t ws_size,
                              hipStream_t stream) {
    (void)in_sizes; (void)n_in; (void)out_size; (void)ws_size;
    // input order = setup_inputs() dict order
    const float* loc_p[3]  = { (const float*)d_in[0], (const float*)d_in[4], (const float*)d_in[8] };
    const float* cls_p[3]  = { (const float*)d_in[1], (const float*)d_in[5], (const float*)d_in[9] };
    const float* box_p[3]  = { (const float*)d_in[2], (const float*)d_in[6], (const float*)d_in[10] };
    const float* ctr_p[3]  = { (const float*)d_in[3], (const float*)d_in[7], (const float*)d_in[11] };
    const float* can_pred  = (const float*)d_in[13];
    const float* ins_pred  = (const float*)d_in[14];
    const int*   img_h     = (const int*)d_in[15];
    const int*   img_w     = (const int*)d_in[16];

    const int Nlvl[3]   = { 128 * 128, 64 * 64, 32 * 32 };
    const int acc[3]    = { 0, 16384, 20480 };
    const size_t mOff[3] = { 0, 16384ull * NCLS, 20480ull * NCLS };  // elements
    const size_t mTotal  = 21504ull * NCLS;

    char* ws = (char*)d_ws;
    size_t o = 0;
    float*    masked   = (float*)(ws + o);   o = alignup(o + mTotal * 4, 256);
    unsigned* hist     = (unsigned*)(ws + o); o = alignup(o + 3 * 2048 * 4, 256);
    unsigned* cutKey   = (unsigned*)(ws + o); o = alignup(o + 3 * 4, 256);
    unsigned* counters = (unsigned*)(ws + o); o = alignup(o + 3 * 4, 256);
    float*    mScores  = (float*)(ws + o);   o = alignup(o + M3 * 4, 256);
    float*    mBoxes   = (float*)(ws + o);   o = alignup(o + M3 * 16, 256);
    int*      mLabels  = (int*)(ws + o);     o = alignup(o + M3 * 4, 256);
    int*      mLocs    = (int*)(ws + o);     o = alignup(o + M3 * 4, 256);
    int*      keep     = (int*)(ws + o);     o = alignup(o + M3 * 4, 256);
    int*      locp     = (int*)(ws + o);     o = alignup(o + 100 * 4, 256);
    int*      okf      = (int*)(ws + o);     o = alignup(o + 100 * 4, 256);
    _Float16* ins16    = (_Float16*)(ws + o); o = alignup(o + KPAD * EMB * 2, 256);
    _Float16* canT     = (_Float16*)(ws + o); o = alignup(o + (size_t)HWSEG * EMB * 2, 256);
    float*    seg      = (float*)(ws + o);   o = alignup(o + (size_t)KPAD * HWSEG * 4, 256);

    float* out = (float*)d_out;

    // 0) init
    init_kernel<<<(M3 * 4 + 255) / 256, 256, 0, stream>>>(hist, counters, mScores, mBoxes,
                                                          mLabels, mLocs, keep);
    // 1) score + hist (per level), 2) cutoff, 3) compact
    for (int l = 0; l < 3; ++l)
        score_hist_kernel<<<512, 256, 0, stream>>>(cls_p[l], ctr_p[l], masked + mOff[l],
                                                   hist + l * 2048, Nlvl[l]);
    cutoff_kernel<<<1, 64, 0, stream>>>(hist, cutKey);
    for (int l = 0; l < 3; ++l) {
        int total = Nlvl[l] * NCLS;
        compact_kernel<<<(total + 255) / 256, 256, 0, stream>>>(
            masked + mOff[l], box_p[l], loc_p[l], cutKey, counters, l,
            mScores, mBoxes, mLabels, mLocs, Nlvl[l], acc[l], img_h, img_w);
    }
    // 4) NMS over classes 1..80
    nms_kernel<<<80, 256, 0, stream>>>(mScores, mBoxes, mLabels, keep);
    // 5) final top-100 + scalar outputs
    final_kernel<<<1, 256, 0, stream>>>(mScores, mBoxes, mLabels, mLocs, keep, out, locp, okf);
    // 6) f16 conversion / gather
    convert_can_kernel<<<HWSEG / 64, 256, 0, stream>>>(can_pred, canT);
    gather_ins_kernel<<<(KPAD * EMB + 255) / 256, 256, 0, stream>>>(ins_pred, locp, okf, ins16);
    // 7) WMMA mask-logit GEMM: 7 M-tiles x 1024 N-tile groups, 8 waves/block
    {
        int nWaves = (KPAD / 16) * (HWSEG / 16 / NTPW);   // 7168
        wmma_gemm_kernel<<<(nWaves * 32 + 255) / 256, 256, 0, stream>>>(ins16, canT, seg);
    }
    // 8) fused upsample + threshold (419 MB streaming NT write)
    {
        long long tot4 = 100LL * 1024 * 256;
        resize_kernel<<<(unsigned)((tot4 + 255) / 256), 256, 0, stream>>>(seg, okf, out);
    }
}
